// VanillaGNNLayer_5600637354090
// MI455X (gfx1250) — compile-verified, matrix operand order checked by reading, not run on hardware
//
#include <hip/hip_runtime.h>

typedef __attribute__((ext_vector_type(16))) __bf16 v16bf;
typedef __attribute__((ext_vector_type(8)))  __bf16 v8bf;
typedef __attribute__((ext_vector_type(8)))  float  v8f;

#define GNN_N_NODES 100000
#define GNN_DIM     256

// ---------------------------------------------------------------------------
// Zero d_out (harness poisons it with 0xAA; atomics need a clean zero base).
// ---------------------------------------------------------------------------
__global__ __launch_bounds__(256) void gnn_zero_kernel(float4* __restrict__ p, long long n4) {
    long long i = (long long)blockIdx.x * blockDim.x + threadIdx.x;
    if (i < n4) p[i] = make_float4(0.f, 0.f, 0.f, 0.f);
}

// ---------------------------------------------------------------------------
// GEMM: h[n][o] = sum_k x[n][k] * W[o][k]   (h = x @ W^T).
// bf16 WMMA with f32 accumulation (v_wmma_f32_16x16x32_bf16); h stored bf16
// so the edge-gather working set (h 51MB + out 102MB) fits in the 192MB L2.
//
// Block = 256 threads = 8 waves. Block computes 64 rows x 256 cols of h.
//   wave w: rows rbase = blockIdx.x*64 + (w&3)*16, cols cbase = (w>>2)*128
//   -> 8 accumulator tiles of 16x16 per wave.
//
// 16-bit A fragment layout (ISA 7.12.2): lane L holds row M = L&15;
//   elements 0..7  = K in [k0 + (L>>4)*8,      +8)
//   elements 8..15 = K in [k0 + 16 + (L>>4)*8, +8)
// B (32x16) mirrors this with lane = output column = W row index, so both
// operands load with the identical per-lane pattern from row-major storage.
// ---------------------------------------------------------------------------
__global__ __launch_bounds__(256) void gnn_gemm_kernel(const float* __restrict__ x,
                                                       const float* __restrict__ W,
                                                       __bf16* __restrict__ h) {
    const int lane  = threadIdx.x & 31;
    const int w     = threadIdx.x >> 5;
    const int m     = lane & 15;        // row within tile (A) / col within tile (B)
    const int koff  = (lane >> 4) * 8;  // K sub-offset for upper half-wave
    const int rbase = blockIdx.x * 64 + (w & 3) * 16;
    const int cbase = (w >> 2) * 128;

    int arow = rbase + m;
    if (arow >= GNN_N_NODES) arow = 0;  // clamp loads; stores are guarded
    const float* xr = x + (size_t)arow * GNN_DIM;

    v8f acc[8] = {};

    for (int k0 = 0; k0 < GNN_DIM; k0 += 32) {
        // ---- A fragment: 16 floats -> v16bf
        float af[16];
        {
            const float4* p0 = (const float4*)(xr + k0 + koff);
            const float4* p1 = (const float4*)(xr + k0 + 16 + koff);
            *(float4*)(af + 0)  = p0[0];
            *(float4*)(af + 4)  = p0[1];
            *(float4*)(af + 8)  = p1[0];
            *(float4*)(af + 12) = p1[1];
        }
        v16bf a;
#pragma unroll
        for (int i = 0; i < 16; ++i) a[i] = (__bf16)af[i];

        // ---- 8 column tiles: load B fragment from W, one WMMA each
#pragma unroll
        for (int t = 0; t < 8; ++t) {
            const float* wr = W + (size_t)(cbase + t * 16 + m) * GNN_DIM;
            float bf[16];
            {
                const float4* q0 = (const float4*)(wr + k0 + koff);
                const float4* q1 = (const float4*)(wr + k0 + 16 + koff);
                *(float4*)(bf + 0)  = q0[0];
                *(float4*)(bf + 4)  = q0[1];
                *(float4*)(bf + 8)  = q1[0];
                *(float4*)(bf + 12) = q1[1];
            }
            v16bf b;
#pragma unroll
            for (int i = 0; i < 16; ++i) b[i] = (__bf16)bf[i];

            acc[t] = __builtin_amdgcn_wmma_f32_16x16x32_bf16(
                /*neg_a=*/false, a, /*neg_b=*/false, b,
                /*c_mod=*/(short)0, acc[t], /*reuse_a=*/false, /*reuse_b=*/false);
        }
    }

    // ---- store (bf16): C/D layout (ISA 7.12.2): lane 0-15 -> N=lane, VGPR v -> M=v;
    //                     lane 16-31 -> N=lane-16, M=v+8
#pragma unroll
    for (int t = 0; t < 8; ++t) {
        const int col = cbase + t * 16 + m;
#pragma unroll
        for (int v = 0; v < 8; ++v) {
            const int row = rbase + v + (lane >> 4) * 8;
            if (row < GNN_N_NODES) h[(size_t)row * GNN_DIM + col] = (__bf16)acc[t][v];
        }
    }
}

// ---------------------------------------------------------------------------
// Edge scatter: out[rows[e]] += vals[e] * h[cols[e]]  (256 elems per row).
// One wave32 per edge: lane covers 8 consecutive bf16 of the gathered row
// (one global_load_b128 = 16 bytes), converts to f32, then eight
// non-returning global_atomic_add_f32 into the L2-resident output.
// ---------------------------------------------------------------------------
__global__ __launch_bounds__(256) void gnn_edge_kernel(const __bf16* __restrict__ h,
                                                       const int* __restrict__ rows,
                                                       const int* __restrict__ cols,
                                                       const float* __restrict__ vals,
                                                       float* __restrict__ out,
                                                       int n_edges) {
    const int lane = threadIdx.x & 31;
    const long long e = ((long long)blockIdx.x * blockDim.x + threadIdx.x) >> 5;
    if (e >= n_edges) return;

    const int   r = rows[e];
    const int   c = cols[e];
    const float v = vals[e];

    const v8bf* hr = (const v8bf*)(h + (size_t)c * GNN_DIM);
    v8bf hv = hr[lane];                       // 8 bf16 = one b128 load

    float* orow = out + (size_t)r * GNN_DIM + 8 * lane;
#pragma unroll
    for (int i = 0; i < 8; ++i) {
        atomicAdd(orow + i, v * (float)hv[i]);
    }
}

// ---------------------------------------------------------------------------
// Launch: inputs = { x, adj_rows, adj_cols, adj_vals, W }
// d_ws holds h = x @ W^T in bf16  (100000 * 256 * 2B = 51.2 MB)
// ---------------------------------------------------------------------------
extern "C" void kernel_launch(void* const* d_in, const int* in_sizes, int n_in,
                              void* d_out, int out_size, void* d_ws, size_t ws_size,
                              hipStream_t stream) {
    const float* x        = (const float*)d_in[0];
    const int*   adj_rows = (const int*)d_in[1];
    const int*   adj_cols = (const int*)d_in[2];
    const float* adj_vals = (const float*)d_in[3];
    const float* W        = (const float*)d_in[4];

    __bf16* h   = (__bf16*)d_ws;
    float*  out = (float*)d_out;
    const int n_edges = in_sizes[1];

    // 1) zero the output accumulator
    {
        long long n4 = (long long)out_size / 4;
        int blocks = (int)((n4 + 255) / 256);
        gnn_zero_kernel<<<blocks, 256, 0, stream>>>((float4*)d_out, n4);
    }

    // 2) h = x @ W^T via bf16 WMMA (64 rows x 256 cols per block)
    {
        int blocks = (GNN_N_NODES + 63) / 64;
        gnn_gemm_kernel<<<blocks, 256, 0, stream>>>(x, W, h);
    }

    // 3) edge scatter-sum with f32 atomics (one wave per edge)
    {
        int blocks = (n_edges + 7) / 8;  // 8 waves per 256-thread block
        gnn_edge_kernel<<<blocks, 256, 0, stream>>>(h, adj_rows, adj_cols, adj_vals, out, n_edges);
    }
}
